// NonLocalBlock_24962349925127
// MI455X (gfx1250) — matrix operand order repaired and checked by reading, not compiled, not run on hardware
//
#include <hip/hip_runtime.h>

#define BB    8
#define CC    512
#define NN    4096
#define CRED  128
#define OPROJ 384

typedef __bf16 bf16;
typedef __attribute__((ext_vector_type(4)))  int   v4i;
typedef __attribute__((ext_vector_type(8)))  float v8f;
typedef __attribute__((ext_vector_type(8)))  bf16  v8bf;
typedef __attribute__((ext_vector_type(16))) bf16  v16bf;

// ---- async global->LDS (ASYNCcnt path), builtin if present, else inline asm ----
#if defined(__has_builtin) && __has_builtin(__builtin_amdgcn_global_load_async_to_lds_b128)
#define ASYNC_B128(gp, lp)                                                        \
  __builtin_amdgcn_global_load_async_to_lds_b128(                                 \
      (__attribute__((address_space(1))) v4i*)(gp),                               \
      (__attribute__((address_space(3))) v4i*)(lp), 0, 0)
#else
#define ASYNC_B128(gp, lp)                                                        \
  do {                                                                            \
    unsigned _ldsoff =                                                            \
        (unsigned)(unsigned long long)(__attribute__((address_space(3))) void*)(lp); \
    asm volatile("global_load_async_to_lds_b128 %0, %1, off" ::"v"(_ldsoff),      \
                 "v"((const void*)(gp))                                           \
                 : "memory");                                                     \
  } while (0)
#endif

#if defined(__has_builtin) && __has_builtin(__builtin_amdgcn_s_wait_asynccnt)
#define WAIT_ASYNC(n) __builtin_amdgcn_s_wait_asynccnt(n)
#else
#define WAIT_ASYNC(n) asm volatile("s_wait_asynccnt %0" ::"n"(n))
#endif

static __device__ __forceinline__ bf16 f2bf(float f) {
  unsigned u = __builtin_bit_cast(unsigned, f);
  u += 0x7FFFu + ((u >> 16) & 1u);          // round-to-nearest-even
  unsigned short s = (unsigned short)(u >> 16);
  return __builtin_bit_cast(bf16, s);
}

static __device__ __forceinline__ v16bf comb(v8bf lo, v8bf hi) {
  return __builtin_shufflevector(lo, hi, 0,1,2,3,4,5,6,7,8,9,10,11,12,13,14,15);
}

// A fragment (16x32 bf16): per lane, two contiguous runs of 8 halves at +0 and +16
static __device__ __forceinline__ v16bf ldA(const bf16* p) {
  v8bf lo = *(const v8bf*)p;
  v8bf hi = *(const v8bf*)(p + 16);
  return comb(lo, hi);
}
// B fragment (32x16 bf16): per lane, two contiguous runs of 8 halves at +0 and +16
static __device__ __forceinline__ v16bf ldB(const bf16* p) {
  v8bf lo = *(const v8bf*)p;
  v8bf hi = *(const v8bf*)(p + 16);
  return comb(lo, hi);
}

static __device__ __forceinline__ v8f wmma_bf16(v16bf a, v16bf b, v8f c) {
  return __builtin_amdgcn_wmma_f32_16x16x32_bf16(false, a, false, b, (short)0, c,
                                                 false, false);
}

// ---------------------------------------------------------------------------
// Kernel 0: convert weights f32 -> bf16 once.
__global__ __launch_bounds__(256) void wconv_kernel(
    const float* __restrict__ w_in, const float* __restrict__ w_out,
    bf16* __restrict__ wi, bf16* __restrict__ wo) {
  int i = blockIdx.x * 256 + threadIdx.x;
  if (i < OPROJ * CC) wi[i] = f2bf(w_in[i]);
  if (i < CC * CRED)  wo[i] = f2bf(w_out[i]);
}

// ---------------------------------------------------------------------------
// Kernel 1: proj = w_in(384x512) @ x(512xN) per batch via WMMA.
// Writes Q,K token-major [B][N][CR] bf16, V channel-major [B][CR][N] bf16.
__global__ __launch_bounds__(256) void proj_kernel(
    const float* __restrict__ x, const bf16* __restrict__ wi,
    bf16* __restrict__ Qt, bf16* __restrict__ Kt, bf16* __restrict__ Vc) {
  const int b   = blockIdx.y;
  const int n0  = blockIdx.x * 16;
  const int tid = threadIdx.x;
  const int w   = tid >> 5;
  const int l   = tid & 31;
  const int lrow = l & 15;
  const int hi16 = l >> 4;
  const int aoff = hi16 << 3;   // 8*(lane>=16)
  const int boff = hi16 << 4;   // 16*(lane>=16)

  __shared__ __align__(16) bf16 xsT[16 * 40];   // [n][c] tile, padded stride 40
  const float* xb = x + (size_t)b * CC * NN;

  v8f acc[3];
#pragma unroll
  for (int t = 0; t < 3; ++t) acc[t] = (v8f){0,0,0,0,0,0,0,0};

  for (int kc = 0; kc < 16; ++kc) {
    const int c0 = kc * 32;
    __syncthreads();
#pragma unroll
    for (int e = 0; e < 2; ++e) {             // stage x[c0..c0+31][n0..n0+15] -> bf16
      int idx = tid + e * 256;
      int ccl = idx >> 4;
      int nnl = idx & 15;
      xsT[nnl * 40 + ccl] = f2bf(xb[(size_t)(c0 + ccl) * NN + n0 + nnl]);
    }
    __syncthreads();
    v16bf bfr = ldB(&xsT[lrow * 40 + aoff]);  // B: rows c, cols n
#pragma unroll
    for (int t = 0; t < 3; ++t) {
      const int o0 = (w * 3 + t) * 16;
      v16bf afr = ldA(wi + (size_t)(o0 + lrow) * CC + c0 + aoff);  // A: rows o, cols c
      acc[t] = wmma_bf16(afr, bfr, acc[t]);
    }
  }
#pragma unroll
  for (int t = 0; t < 3; ++t) {
    const int o0 = (w * 3 + t) * 16;
#pragma unroll
    for (int r = 0; r < 8; ++r) {
      const int oo = o0 + r + (hi16 << 3);
      const int n  = n0 + lrow;
      bf16 v = f2bf(acc[t][r]);
      if (o0 < 128)       Qt[((size_t)b * NN + n) * CRED + oo] = v;
      else if (o0 < 256)  Kt[((size_t)b * NN + n) * CRED + (oo - 128)] = v;
      else                Vc[((size_t)b * CRED + (oo - 256)) * NN + n] = v;
    }
  }
}

// ---------------------------------------------------------------------------
// Stage one 32-column K/V slab into LDS via async global->LDS DMA.
// K-slab: [32 m][136 c-stride] bf16 ; V-slab: [128 c][40 m-stride] bf16.
// Per thread: 4 + 4 b128 ops (per wave: 8 instructions -> ASYNCcnt 8).
static __device__ __forceinline__ void stage_slab(const bf16* __restrict__ Kb,
                                                  const bf16* __restrict__ Vb,
                                                  int m0, bf16* kd, bf16* vd,
                                                  int tid) {
#pragma unroll
  for (int i = 0; i < 4; ++i) {
    const int g  = tid + i * 128;
    const int kr = g >> 4, kch = g & 15;      // 32 rows x 16 chunks of 16B
    ASYNC_B128(Kb + (size_t)(m0 + kr) * CRED + kch * 8, kd + kr * 136 + kch * 8);
    const int vr = g >> 2, vch = g & 3;       // 128 rows x 4 chunks of 16B
    ASYNC_B128(Vb + (size_t)vr * NN + m0 + vch * 8, vd + vr * 40 + vch * 8);
  }
}

// ---------------------------------------------------------------------------
// Kernel 2: fused flash-attention + output projection + residual.
// Block = 128 thr (4 waves); wave owns one 16-row query tile; the block shares
// double-buffered K/V slabs staged by async DMA.
__global__ __launch_bounds__(128) void attn_kernel(
    const float* __restrict__ x, const bf16* __restrict__ Qt,
    const bf16* __restrict__ Kt, const bf16* __restrict__ Vc,
    const bf16* __restrict__ wo, float* __restrict__ out) {
  const int b   = blockIdx.y;
  const int tid = threadIdx.x;
  const int w   = tid >> 5;
  const int l   = tid & 31;
  const int n0   = (blockIdx.x * 4 + w) * 16;
  const int lrow = l & 15;
  const int hi16 = l >> 4;
  const int aoff = hi16 << 3;
  const int boff = hi16 << 4;

  __shared__ __align__(16) bf16 kslab[2 * 32 * 136];   // 17.0 KB
  __shared__ __align__(16) bf16 vslab[2 * 128 * 40];   // 20.0 KB
  __shared__ __align__(16) bf16 pscr[4 * 16 * 136];    // 17.0 KB (per-wave P / O^T)
  bf16* lsb = pscr + w * (16 * 136);

  // Q A-fragments, resident for the whole pass
  const bf16* qrow = Qt + ((size_t)b * NN + (n0 + lrow)) * CRED;
  v16bf qa[4];
#pragma unroll
  for (int kc = 0; kc < 4; ++kc) qa[kc] = ldA(qrow + kc * 32 + aoff);

  v8f o[8];
#pragma unroll
  for (int ct = 0; ct < 8; ++ct) o[ct] = (v8f){0,0,0,0,0,0,0,0};
  float mrun[8], lrun[8];
#pragma unroll
  for (int r = 0; r < 8; ++r) { mrun[r] = -3.0e38f; lrun[r] = 0.0f; }

  const bf16* Kb = Kt + (size_t)b * NN * CRED;
  const bf16* Vb = Vc + (size_t)b * CRED * NN;

  stage_slab(Kb, Vb, 0, kslab, vslab, tid);            // prologue: slab 0

  for (int ms = 0; ms < NN / 32; ++ms) {
    const int m0 = ms * 32;
    bf16* kcur = kslab + (ms & 1) * (32 * 136);
    bf16* vcur = vslab + (ms & 1) * (128 * 40);
    if (ms + 1 < NN / 32) {                            // prefetch next slab
      stage_slab(Kb, Vb, m0 + 32,
                 kslab + ((ms + 1) & 1) * (32 * 136),
                 vslab + ((ms + 1) & 1) * (128 * 40), tid);
      WAIT_ASYNC(8);                                   // slab ms complete (in-order)
    } else {
      WAIT_ASYNC(0);
    }
    __syncthreads();                                   // slab ms visible to all waves

    v8f s0 = (v8f){0,0,0,0,0,0,0,0};
    v8f s1 = (v8f){0,0,0,0,0,0,0,0};
#pragma unroll
    for (int kc = 0; kc < 4; ++kc) {                   // S = Q^T K  (K-dim = 128)
      v16bf k0 = ldB(kcur + (lrow)      * 136 + kc * 32 + aoff);
      v16bf k1 = ldB(kcur + (16 + lrow) * 136 + kc * 32 + aoff);
      s0 = wmma_bf16(qa[kc], k0, s0);
      s1 = wmma_bf16(qa[kc], k1, s1);
    }
    // online softmax over this 32-column slab; row = 16-lane group reduction
#pragma unroll
    for (int r = 0; r < 8; ++r) {
      float a0 = s0[r], a1 = s1[r];
      float mx = fmaxf(a0, a1);
      mx = fmaxf(mx, __shfl_xor(mx, 1, 32));
      mx = fmaxf(mx, __shfl_xor(mx, 2, 32));
      mx = fmaxf(mx, __shfl_xor(mx, 4, 32));
      mx = fmaxf(mx, __shfl_xor(mx, 8, 32));
      float nm   = fmaxf(mrun[r], mx);
      float corr = __expf(mrun[r] - nm);
      float p0   = __expf(a0 - nm);
      float p1   = __expf(a1 - nm);
      float psum = p0 + p1;
      psum += __shfl_xor(psum, 1, 32);
      psum += __shfl_xor(psum, 2, 32);
      psum += __shfl_xor(psum, 4, 32);
      psum += __shfl_xor(psum, 8, 32);
      lrun[r] = lrun[r] * corr + psum;
      mrun[r] = nm;
#pragma unroll
      for (int ct = 0; ct < 8; ++ct) o[ct][r] *= corr;
      const int prow = r + (hi16 << 3);
      lsb[prow * 40 + lrow]      = f2bf(p0);           // C-layout -> [n][m] tile
      lsb[prow * 40 + 16 + lrow] = f2bf(p1);
    }
    // re-read P in A-layout (same-wave LDS RAW; DScnt handled by compiler)
    const bf16* pr = lsb + lrow * 40 + aoff;
    v16bf pa = comb(*(const v8bf*)pr, *(const v8bf*)(pr + 16));
#pragma unroll
    for (int ct = 0; ct < 8; ++ct) {                   // O += P V
      v16bf vb = ldB(vcur + (ct * 16 + lrow) * 40 + aoff);
      o[ct] = wmma_bf16(pa, vb, o[ct]);
    }
    __syncthreads();                                   // all waves done with slab ms
  }

  // normalize and park O^T (as [n][c] bf16) in LDS for the output projection
#pragma unroll
  for (int ct = 0; ct < 8; ++ct)
#pragma unroll
    for (int r = 0; r < 8; ++r) {
      const int nrow = r + (hi16 << 3);
      lsb[nrow * 136 + ct * 16 + lrow] = f2bf(o[ct][r] / lrun[r]);
    }

  v16bf ob[4];
#pragma unroll
  for (int kc = 0; kc < 4; ++kc)                       // B: rows c (K=128), cols n
    ob[kc] = ldB(lsb + lrow * 136 + kc * 32 + aoff);

  for (int ot = 0; ot < 32; ++ot) {                    // y = w_out @ O^T, + residual
    v8f acc = (v8f){0,0,0,0,0,0,0,0};
#pragma unroll
    for (int kc = 0; kc < 4; ++kc) {
      v16bf aw = ldA(wo + (size_t)(ot * 16 + lrow) * CRED + kc * 32 + aoff);
      acc = wmma_bf16(aw, ob[kc], acc);
    }
#pragma unroll
    for (int r = 0; r < 8; ++r) {
      const int cp = ot * 16 + r + (hi16 << 3);
      const size_t idx = ((size_t)b * CC + cp) * NN + n0 + lrow;
      out[idx] = x[idx] + acc[r];
    }
  }
}

// ---------------------------------------------------------------------------
extern "C" void kernel_launch(void* const* d_in, const int* in_sizes, int n_in,
                              void* d_out, int out_size, void* d_ws, size_t ws_size,
                              hipStream_t stream) {
  (void)in_sizes; (void)n_in; (void)out_size; (void)ws_size;
  const float* x     = (const float*)d_in[0];
  const float* w_in  = (const float*)d_in[1];
  const float* w_out = (const float*)d_in[2];
  float* out = (float*)d_out;

  bf16* Qt = (bf16*)d_ws;                                // [B][N][CR]  8 MB
  bf16* Kt = Qt + (size_t)BB * NN * CRED;                // [B][N][CR]  8 MB
  bf16* Vc = Kt + (size_t)BB * NN * CRED;                // [B][CR][N]  8 MB
  bf16* wi = Vc + (size_t)BB * NN * CRED;                // [384][512]
  bf16* wo = wi + (size_t)OPROJ * CC;                    // [512][128]

  wconv_kernel<<<(OPROJ * CC + 255) / 256, 256, 0, stream>>>(w_in, w_out, wi, wo);
  proj_kernel<<<dim3(NN / 16, BB), 256, 0, stream>>>(x, wi, Qt, Kt, Vc);
  attn_kernel<<<dim3(NN / 16 / 4, BB), 128, 0, stream>>>(x, Qt, Kt, Vc, wo, out);
}
